// Encoder_5626407158302
// MI455X (gfx1250) — matrix operand order repaired and checked by reading, not compiled, not run on hardware
//
#include <hip/hip_runtime.h>

// ---------------------------------------------------------------------------
// Problem constants (from reference)
// ---------------------------------------------------------------------------
#define BB   4
#define N0_  10000
#define E0_  80000
#define N1_  2500
#define E1_  20000
#define N2_  625
#define E2_  5000
#define HID_ 128
#define LAT_ 64

typedef __bf16 bf16x16 __attribute__((ext_vector_type(16)));
typedef float  f32x8   __attribute__((ext_vector_type(8)));

// ---------------------------------------------------------------------------
// bf16 helpers (storage = unsigned short)
// ---------------------------------------------------------------------------
__device__ __forceinline__ unsigned short f2bf(float f) {
  union { float f; unsigned int u; } v; v.f = f;
  unsigned int u = v.u;
  u += 0x7FFFu + ((u >> 16) & 1u);   // round-to-nearest-even
  return (unsigned short)(u >> 16);
}
__device__ __forceinline__ float bf2f(unsigned short h) {
  union { unsigned int u; float f; } v; v.u = ((unsigned int)h) << 16;
  return v.f;
}

// ---------------------------------------------------------------------------
// Weight packing: W (K x N, f32, row-major) -> CDNA5 WMMA B-fragment layout.
// Fragment layout used here (contiguous k, same as A) (16-bit B, 32x16, wave32):
//   VGPR j (j=0..7), lane l : N = 16*ntile + (l & 15), K = 32*ks + 16*(l>>4) + 2j (+1)
// Packed storage: [ntile][kstep][lane][16 bf16]  (32B per lane -> two b128 loads)
// ---------------------------------------------------------------------------
__global__ void pack_w_kernel(const float* __restrict__ W, unsigned short* __restrict__ out,
                              int K, int N, int Kpad) {
  int Ksteps = Kpad >> 5;
  long total = (long)(N >> 4) * Ksteps * 32;
  long idx = (long)blockIdx.x * blockDim.x + threadIdx.x;
  if (idx >= total) return;
  int lane = (int)(idx & 31);
  long t = idx >> 5;
  int ks = (int)(t % Ksteps);
  int nt = (int)(t / Ksteps);
  int n = nt * 16 + (lane & 15);
  int h = lane >> 4;
  alignas(16) unsigned short vals[16];
#pragma unroll
  for (int j = 0; j < 8; ++j) {
    int k0 = ks * 32 + h * 16 + 2 * j;
    vals[2*j]   = (k0     < K) ? f2bf(W[(size_t)k0 * N + n])     : (unsigned short)0;
    vals[2*j+1] = (k0 + 1 < K) ? f2bf(W[(size_t)(k0+1) * N + n]) : (unsigned short)0;
  }
  uint4* dst = (uint4*)(out + idx * 16);
  dst[0] = ((const uint4*)vals)[0];
  dst[1] = ((const uint4*)vals)[1];
}

// ---------------------------------------------------------------------------
// WMMA GEMM:  out[M,N] = act(A[M,Kpad](bf16) * Wpacked + bias)
// Each wave computes a 32x64 strip: two 16-row halves share every B fragment
// (8 v_wmma_f32_16x16x32_bf16 per K-step for 4 A-loads + 8 B-loads).
// All four B fragments are loaded into distinct registers before the WMMA
// burst so the loads pipeline (graduated s_wait_loadcnt instead of wait-0).
// A-fragment layout used here (contiguous k, same as B) (16-bit A, 16x32, wave32):
//   lane l: M = m0 + (l&15); VGPR j: K = 32ks + 16*(l>>4) + 2j
//   -> two contiguous 16B loads per lane from row-major A.
// act: 0 none, 1 relu, 2 leaky_relu(0.01)
// ---------------------------------------------------------------------------
__global__ __launch_bounds__(256)
void wmma_gemm_bf16(const unsigned short* __restrict__ A,
                    const unsigned short* __restrict__ Bp,
                    const float* __restrict__ bias,
                    float* __restrict__ outF,
                    unsigned short* __restrict__ outH,
                    int Mtiles32, int N, int Kpad, int act, int outIsF32) {
  int wave = threadIdx.x >> 5;
  int lane = threadIdx.x & 31;
  int mt = blockIdx.x * 8 + wave;
  if (mt >= Mtiles32) return;
  int ngrp = blockIdx.y;            // group of 4 n-tiles (64 cols)
  int Ksteps = Kpad >> 5;
  int m0 = mt * 32;
  int h = lane >> 4;
  const unsigned short* arow0 = A + (size_t)(m0 + (lane & 15)) * Kpad + h * 16;
  const unsigned short* arow1 = arow0 + (size_t)16 * Kpad;

  f32x8 acc[2][4];
#pragma unroll
  for (int t = 0; t < 4; ++t) {
    int n = (ngrp * 4 + t) * 16 + (lane & 15);
    float bv = bias ? bias[n] : 0.0f;
#pragma unroll
    for (int i = 0; i < 8; ++i) { acc[0][t][i] = bv; acc[1][t][i] = bv; }
  }

  for (int ks = 0; ks < Ksteps; ++ks) {
    union F { uint4 u[2]; bf16x16 v; };
    F a0, a1, bf[4];
    a0.u[0] = *(const uint4*)(arow0 + ks * 32);
    a0.u[1] = *(const uint4*)(arow0 + ks * 32 + 8);
    a1.u[0] = *(const uint4*)(arow1 + ks * 32);
    a1.u[1] = *(const uint4*)(arow1 + ks * 32 + 8);
#pragma unroll
    for (int t = 0; t < 4; ++t) {
      int nt = ngrp * 4 + t;
      const uint4* bp = (const uint4*)(Bp + (((size_t)nt * Ksteps + ks) * 32 + lane) * 16);
      bf[t].u[0] = bp[0];
      bf[t].u[1] = bp[1];
    }
#pragma unroll
    for (int t = 0; t < 4; ++t) {
      acc[0][t] = __builtin_amdgcn_wmma_f32_16x16x32_bf16(
          false, a0.v, false, bf[t].v, (short)0, acc[0][t], false, false);
      acc[1][t] = __builtin_amdgcn_wmma_f32_16x16x32_bf16(
          false, a1.v, false, bf[t].v, (short)0, acc[1][t], false, false);
    }
  }

#pragma unroll
  for (int p = 0; p < 2; ++p) {
#pragma unroll
    for (int t = 0; t < 4; ++t) {
      int n = (ngrp * 4 + t) * 16 + (lane & 15);
#pragma unroll
      for (int i = 0; i < 8; ++i) {
        float v = acc[p][t][i];
        if (act == 1)      v = v > 0.0f ? v : 0.0f;
        else if (act == 2) v = v > 0.0f ? v : 0.01f * v;
        int r = m0 + p * 16 + h * 8 + i;
        if (outIsF32) outF[(size_t)r * N + n] = v;
        else          outH[(size_t)r * N + n] = f2bf(v);
      }
    }
  }
}

// ---------------------------------------------------------------------------
// LayerNorm over rows of f32 [M,N]; one wave (32 lanes) per row.
// Writes bf16 (outH) and optionally normalized f32 back in place (outF).
// ---------------------------------------------------------------------------
__global__ __launch_bounds__(256)
void layernorm_rows(const float* __restrict__ X, int M, int N,
                    const float* __restrict__ g, const float* __restrict__ be,
                    unsigned short* __restrict__ outH, float* __restrict__ outF) {
  int wave = threadIdx.x >> 5;
  int lane = threadIdx.x & 31;
  int row = blockIdx.x * 8 + wave;
  if (row >= M) return;
  const float* x = X + (size_t)row * N;
  float s = 0.0f, s2 = 0.0f;
  for (int c = lane; c < N; c += 32) { float v = x[c]; s += v; s2 += v * v; }
  for (int off = 16; off; off >>= 1) {
    s  += __shfl_xor(s,  off, 32);
    s2 += __shfl_xor(s2, off, 32);
  }
  float mu = s / (float)N;
  float var = s2 / (float)N - mu * mu;
  float rs = rsqrtf(var + 1e-5f);
  for (int c = lane; c < N; c += 32) {
    float o = (x[c] - mu) * rs * g[c] + be[c];
    outH[(size_t)row * N + c] = f2bf(o);
    if (outF) outF[(size_t)row * N + c] = o;
  }
}

// ---------------------------------------------------------------------------
// Edge concat: A[b*E+i, 0:C]=x[b,src[i]], [C:2C]=x[b,dst[i]], [2C:3C]=e[b,i]
// 8 bf16 (16B) per thread.
// ---------------------------------------------------------------------------
__global__ void edge_concat(const unsigned short* __restrict__ x,
                            const unsigned short* __restrict__ e,
                            const int* __restrict__ src, const int* __restrict__ dst,
                            unsigned short* __restrict__ A,
                            int Bn, int Nn, int En, int C) {
  int segs = (3 * C) >> 3;
  long total = (long)Bn * En * segs;
  long idx = (long)blockIdx.x * blockDim.x + threadIdx.x;
  if (idx >= total) return;
  int sg = (int)(idx % segs);
  long r = idx / segs;
  int i = (int)(r % En);
  int b = (int)(r / En);
  int c8 = sg * 8;
  const unsigned short* sp;
  if (c8 < C)          sp = x + ((size_t)b * Nn + src[i]) * C + c8;
  else if (c8 < 2 * C) sp = x + ((size_t)b * Nn + dst[i]) * C + (c8 - C);
  else                 sp = e + ((size_t)b * En + i) * C + (c8 - 2 * C);
  *(uint4*)(A + ((size_t)b * En + i) * (3 * C) + c8) = *(const uint4*)sp;
}

// Node concat: A[b*N+n, 0:C]=x, [C:C+Cout]=bf16(agg)
__global__ void node_concat(const unsigned short* __restrict__ x,
                            const float* __restrict__ agg,
                            unsigned short* __restrict__ A,
                            int Bn, int Nn, int C, int Cout) {
  int W = C + Cout;
  int segs = W >> 3;
  long total = (long)Bn * Nn * segs;
  long idx = (long)blockIdx.x * blockDim.x + threadIdx.x;
  if (idx >= total) return;
  int sg = (int)(idx % segs);
  long row = idx / segs;
  int c8 = sg * 8;
  unsigned short* dp = A + (size_t)row * W + c8;
  if (c8 < C) {
    *(uint4*)dp = *(const uint4*)(x + (size_t)row * C + c8);
  } else {
    const float* ap = agg + (size_t)row * Cout + (c8 - C);
    alignas(16) unsigned short v[8];
#pragma unroll
    for (int j = 0; j < 8; ++j) v[j] = f2bf(ap[j]);
    *(uint4*)dp = *(const uint4*)v;
  }
}

// Scatter-add e2 (normalized f32) into agg over dst
__global__ void scatter_add(const float* __restrict__ e2, const int* __restrict__ dst,
                            float* __restrict__ agg, int Bn, int Nn, int En, int C) {
  long total = (long)Bn * En * C;
  long idx = (long)blockIdx.x * blockDim.x + threadIdx.x;
  if (idx >= total) return;
  int c = (int)(idx % C);
  long r = idx / C;
  int i = (int)(r % En);
  int b = (int)(r / En);
  atomicAdd(&agg[((size_t)b * Nn + dst[i]) * C + c], e2[((size_t)b * En + i) * C + c]);
}

// Row gather (pooling / edge selection): out[b,j,:] = in[b, idxs[j], :]
__global__ void gather_rows(const unsigned short* __restrict__ in,
                            const int* __restrict__ idxs,
                            unsigned short* __restrict__ out,
                            int Bn, int NnIn, int NnOut, int C) {
  int segs = C >> 3;
  long total = (long)Bn * NnOut * segs;
  long idx = (long)blockIdx.x * blockDim.x + threadIdx.x;
  if (idx >= total) return;
  int sg = (int)(idx % segs);
  long r = idx / segs;
  int j = (int)(r % NnOut);
  int b = (int)(r / NnOut);
  int c8 = sg * 8;
  *(uint4*)(out + ((size_t)b * NnOut + j) * C + c8) =
      *(const uint4*)(in + ((size_t)b * NnIn + idxs[j]) * C + c8);
}

// x = relu(x + xs), bf16 elementwise
__global__ void residual_relu(unsigned short* __restrict__ x,
                              const unsigned short* __restrict__ xs, long n) {
  long i = (long)blockIdx.x * blockDim.x + threadIdx.x;
  if (i >= n) return;
  float v = bf2f(x[i]) + bf2f(xs[i]);
  x[i] = f2bf(v > 0.0f ? v : 0.0f);
}

// Convert [rows,3] f32 -> [rows,32] bf16 (zero-padded K)
__global__ void pad_convert3(const float* __restrict__ in,
                             unsigned short* __restrict__ out, long rows) {
  long r = (long)blockIdx.x * blockDim.x + threadIdx.x;
  if (r >= rows) return;
  alignas(16) unsigned short v[32];
#pragma unroll
  for (int j = 0; j < 32; ++j)
    v[j] = (j < 3) ? f2bf(in[r * 3 + j]) : (unsigned short)0;
  uint4* dst = (uint4*)(out + r * 32);
#pragma unroll
  for (int k = 0; k < 4; ++k) dst[k] = ((const uint4*)v)[k];
}

// Head transpose: A[(b*64+c)*640 + n] = n<625 ? xb[(b*625+n)*64 + c] : 0
__global__ void head_transpose(const unsigned short* __restrict__ xb,
                               unsigned short* __restrict__ A) {
  int idx = blockIdx.x * blockDim.x + threadIdx.x;
  if (idx >= 256 * 640) return;
  int n = idx % 640;
  int rc = idx / 640;
  int b = rc >> 6;
  int c = rc & 63;
  A[idx] = (n < 625) ? xb[((size_t)b * 625 + n) * 64 + c] : (unsigned short)0;
}

__global__ void zero_f32(float* __restrict__ p, long n) {
  long i = (long)blockIdx.x * blockDim.x + threadIdx.x;
  if (i < n) p[i] = 0.0f;
}

// VAE head: mu/lv/z + KL reduction. One block of 256 (B*LAT).
__global__ void vae_head(const float* __restrict__ zx,
                         const float* __restrict__ Wmu, const float* __restrict__ bmu,
                         const float* __restrict__ Wlv, const float* __restrict__ blv,
                         const float* __restrict__ eps, float* __restrict__ out) {
  __shared__ float red[256];
  int t = threadIdx.x;
  const float* row = zx + (size_t)t * 64;
  float mu = bmu[0], lv = blv[0];
  for (int h = 0; h < 64; ++h) { float v = row[h]; mu += v * Wmu[h]; lv += v * Wlv[h]; }
  out[1 + t] = mu + eps[t] * expf(0.5f * lv);
  red[t] = -0.5f * (1.0f + lv - mu * mu - expf(lv));
  __syncthreads();
  for (int s = 128; s; s >>= 1) { if (t < s) red[t] += red[t + s]; __syncthreads(); }
  if (t == 0) out[0] = red[0] * (1.0f / (float)BB);
}

// ---------------------------------------------------------------------------
// Host-side orchestration
// ---------------------------------------------------------------------------
static inline int cdiv(long a, long b) { return (int)((a + b - 1) / b); }
static inline long pad32r(long r) { return (r + 31) & ~31L; }

struct Ctx { char* base; size_t off; };
static void* takeA(Ctx& c, size_t bytes) {
  void* p = c.base + c.off;
  c.off = (c.off + bytes + 255) & ~(size_t)255;
  return p;
}

struct MlpRT {
  const float *be, *g, *W1, *b1, *W2, *b2;
  int K, Kpad, N;
  unsigned short *pw1, *pw2;
};

static void packW(hipStream_t s, const float* W, int K, int N, unsigned short* out) {
  int Kpad = (K + 31) & ~31;
  long total = (long)(N / 16) * (Kpad / 32) * 32;
  pack_w_kernel<<<cdiv(total, 256), 256, 0, s>>>(W, out, K, N, Kpad);
}

static void gemmRun(hipStream_t s, const unsigned short* A, const unsigned short* pw,
                    const float* bias, long Mpad, int N, int Kpad, int act,
                    float* outF, unsigned short* outH) {
  int Mtiles32 = (int)(Mpad / 32);
  dim3 grid(cdiv(Mtiles32, 8), N / 64);
  wmma_gemm_bf16<<<grid, 256, 0, s>>>(A, pw, bias, outF, outH, Mtiles32, N, Kpad, act,
                                      outF != nullptr ? 1 : 0);
}

// _mlp_apply: GEMM1(act) -> GEMM2 -> LN. Returns normalized f32 (if wantF32).
static float* run_mlp(Ctx& t, hipStream_t s, const unsigned short* A, long Mpad,
                      const MlpRT& m, int act, unsigned short* outH, bool wantF32) {
  unsigned short* H = (unsigned short*)takeA(t, (size_t)Mpad * m.N * 2);
  float* P = (float*)takeA(t, (size_t)Mpad * m.N * 4);
  gemmRun(s, A, m.pw1, m.b1, Mpad, m.N, m.Kpad, act, nullptr, H);
  gemmRun(s, H, m.pw2, m.b2, Mpad, m.N, m.N, 0, P, nullptr);
  layernorm_rows<<<cdiv(Mpad, 8), 256, 0, s>>>(P, (int)Mpad, m.N, m.g, m.be, outH,
                                               wantF32 ? P : nullptr);
  return wantF32 ? P : nullptr;
}

// _mpl_apply: edge-MLP -> scatter-agg -> node-MLP
static void run_mpl(Ctx t, hipStream_t s,
                    const unsigned short* x, const unsigned short* e, const int* g,
                    int Bn, int Nn, int En, int C, int Cout,
                    const MlpRT& me, const MlpRT& mn,
                    unsigned short* xOut, unsigned short* eOut) {
  const int* src = g;
  const int* dst = g + En;
  long MePad = pad32r((long)Bn * En);
  long MnPad = pad32r((long)Bn * Nn);

  unsigned short* Ae = (unsigned short*)takeA(t, (size_t)MePad * (3 * C) * 2);
  edge_concat<<<cdiv((long)Bn * En * ((3 * C) >> 3), 256), 256, 0, s>>>(
      x, e, src, dst, Ae, Bn, Nn, En, C);
  float* Pe = run_mlp(t, s, Ae, MePad, me, 1, eOut, true);

  float* agg = (float*)takeA(t, (size_t)MnPad * Cout * 4);
  zero_f32<<<cdiv(MnPad * Cout, 256), 256, 0, s>>>(agg, MnPad * Cout);
  scatter_add<<<cdiv((long)Bn * En * Cout, 256), 256, 0, s>>>(Pe, dst, agg, Bn, Nn, En, Cout);

  unsigned short* An = (unsigned short*)takeA(t, (size_t)MnPad * (C + Cout) * 2);
  node_concat<<<cdiv((long)Bn * Nn * ((C + Cout) >> 3), 256), 256, 0, s>>>(
      x, agg, An, Bn, Nn, C, Cout);
  run_mlp(t, s, An, MnPad, mn, 1, xOut, false);
}

extern "C" void kernel_launch(void* const* d_in, const int* in_sizes, int n_in,
                              void* d_out, int out_size, void* d_ws, size_t ws_size,
                              hipStream_t stream) {
  (void)in_sizes; (void)out_size; (void)ws_size;
  const float* x_in  = (const float*)d_in[0];
  const float* e_in  = (const float*)d_in[1];
  const int* g0 = (const int*)d_in[2];
  const int* g1 = (const int*)d_in[3];
  const int* g2 = (const int*)d_in[4];
  const int* m_id0 = (const int*)d_in[5];
  const int* m_id1 = (const int*)d_in[6];
  const int* ep0 = (const int*)d_in[7];
  const int* ep1 = (const int*)d_in[8];
  const float* eps = (const float*)d_in[9];

  // Params: jax tree-flatten order (dict keys sorted alphabetically).
  // Top: bottom, edge_enc, l0, l1, lv, mlp_node, mu, node_enc
  // _mpl: edge, node;  _mlp: be, g, l1{W,b}, l2{W,b};  _lin: W, b
  int leafSz[102]; int nl = 0;
  auto addLin = [&](int di, int dou) { leafSz[nl++] = di * dou; leafSz[nl++] = dou; };
  auto addMlp = [&](int di, int dou) { leafSz[nl++] = dou; leafSz[nl++] = dou; addLin(di, dou); addLin(dou, dou); };
  auto addMpl = [&](int ci, int co) { addMlp(3 * ci, co); addMlp(ci + co, co); };
  addMpl(512, 64);                                      // bottom
  addMlp(3, 128);                                       // edge_enc
  addMpl(128, 128); addMpl(128, 256); addMpl(128, 256); // l0: m1, m2, skip
  addMpl(256, 256); addMpl(256, 512); addMpl(256, 512); // l1: m1, m2, skip
  addLin(64, 1);                                        // lv
  addLin(625, 64);                                      // mlp_node
  addLin(64, 1);                                        // mu
  addMlp(3, 128);                                       // node_enc
  size_t pre[103]; pre[0] = 0;
  for (int i = 0; i < 102; ++i) pre[i + 1] = pre[i] + (size_t)leafSz[i];
  bool split = (n_in > 11);
  const float* LP[102];
  for (int i = 0; i < 102; ++i)
    LP[i] = split ? (const float*)d_in[10 + i] : ((const float*)d_in[10]) + pre[i];

  Ctx ar{(char*)d_ws, 0};
  auto mkMlp = [&](int base, int K, int N) {
    MlpRT m;
    m.be = LP[base]; m.g = LP[base + 1];
    m.W1 = LP[base + 2]; m.b1 = LP[base + 3];
    m.W2 = LP[base + 4]; m.b2 = LP[base + 5];
    m.K = K; m.Kpad = (K + 31) & ~31; m.N = N;
    m.pw1 = (unsigned short*)takeA(ar, (size_t)m.Kpad * N * 2);
    m.pw2 = (unsigned short*)takeA(ar, (size_t)N * N * 2);
    packW(stream, m.W1, K, N, m.pw1);
    packW(stream, m.W2, N, N, m.pw2);
    return m;
  };

  MlpRT botE = mkMlp(0, 1536, 64),  botN = mkMlp(6, 576, 64);
  MlpRT encE = mkMlp(12, 3, 128);
  MlpRT l0m1e = mkMlp(18, 384, 128), l0m1n = mkMlp(24, 256, 128);
  MlpRT l0m2e = mkMlp(30, 384, 256), l0m2n = mkMlp(36, 384, 256);
  MlpRT l0ske = mkMlp(42, 384, 256), l0skn = mkMlp(48, 384, 256);
  MlpRT l1m1e = mkMlp(54, 768, 256), l1m1n = mkMlp(60, 512, 256);
  MlpRT l1m2e = mkMlp(66, 768, 512), l1m2n = mkMlp(72, 768, 512);
  MlpRT l1ske = mkMlp(78, 768, 512), l1skn = mkMlp(84, 768, 512);
  const float *Wlv = LP[90], *blv = LP[91];
  const float *Wmn = LP[92], *bmn = LP[93];
  const float *Wmu = LP[94], *bmu = LP[95];
  MlpRT encN = mkMlp(96, 3, 128);
  unsigned short* pwHead = (unsigned short*)takeA(ar, (size_t)640 * 64 * 2);
  packW(stream, Wmn, 625, 64, pwHead);

  // Persistent activations (rows padded to 32 for the 32-row wave tiles)
  long rN0 = (long)BB * N0_, rE0 = (long)BB * E0_;          // 40000, 320000 (mult of 32)
  long rN1 = (long)BB * N1_, rE1 = (long)BB * E1_;          // 10000, 80000
  long rN1p = pad32r(rN1);                                  // 10016
  long rN2p = pad32r((long)BB * N2_), rE2 = (long)BB * E2_; // 2528, 20000
  unsigned short* x0  = (unsigned short*)takeA(ar, (size_t)rN0 * 128 * 2);
  unsigned short* e0  = (unsigned short*)takeA(ar, (size_t)rE0 * 128 * 2);
  unsigned short* x1  = (unsigned short*)takeA(ar, (size_t)rN1p * 256 * 2);
  unsigned short* e1  = (unsigned short*)takeA(ar, (size_t)rE1 * 256 * 2);
  unsigned short* x2  = (unsigned short*)takeA(ar, (size_t)rN2p * 512 * 2);
  unsigned short* e2b = (unsigned short*)takeA(ar, (size_t)rE2 * 512 * 2);
  unsigned short* xb  = (unsigned short*)takeA(ar, (size_t)rN2p * 64 * 2);
  unsigned short* xs0 = (unsigned short*)takeA(ar, (size_t)rN1p * 256 * 2);
  unsigned short* xs1 = (unsigned short*)takeA(ar, (size_t)rN2p * 512 * 2);
  unsigned short* xg0 = (unsigned short*)takeA(ar, (size_t)rN1p * 128 * 2);
  unsigned short* eg0 = (unsigned short*)takeA(ar, (size_t)rE1 * 128 * 2);
  unsigned short* x1g = (unsigned short*)takeA(ar, (size_t)rN1p * 128 * 2);
  unsigned short* e1g = (unsigned short*)takeA(ar, (size_t)rE1 * 128 * 2);
  unsigned short* xg1 = (unsigned short*)takeA(ar, (size_t)rN2p * 256 * 2);
  unsigned short* eg1 = (unsigned short*)takeA(ar, (size_t)rE2 * 256 * 2);
  unsigned short* x2g = (unsigned short*)takeA(ar, (size_t)rN2p * 256 * 2);
  unsigned short* e2g = (unsigned short*)takeA(ar, (size_t)rE2 * 256 * 2);
  unsigned short* junkE = (unsigned short*)takeA(ar, (size_t)rE1 * 256 * 2); // discarded e outs
  unsigned short* Ah  = (unsigned short*)takeA(ar, (size_t)256 * 640 * 2);
  float* zx = (float*)takeA(ar, (size_t)256 * 64 * 4);
  Ctx temp0 = ar;  // all stage temporaries reuse this region

  // ---- Encoders ----
  {
    Ctx t = temp0;
    unsigned short* Ax = (unsigned short*)takeA(t, (size_t)rN0 * 32 * 2);
    pad_convert3<<<cdiv(rN0, 256), 256, 0, stream>>>(x_in, Ax, rN0);
    run_mlp(t, stream, Ax, rN0, encN, 2 /*leaky_relu*/, x0, false);
    unsigned short* Ae = (unsigned short*)takeA(t, (size_t)rE0 * 32 * 2);
    pad_convert3<<<cdiv(rE0, 256), 256, 0, stream>>>(e_in, Ae, rE0);
    run_mlp(t, stream, Ae, rE0, encE, 1 /*relu*/, e0, false);
  }

  // ---- _res_down level 0 (g0 -> g1) ----
  gather_rows<<<cdiv((long)BB * N1_ * 16, 256), 256, 0, stream>>>(x0, m_id0, xg0, BB, N0_, N1_, 128);
  gather_rows<<<cdiv((long)BB * E1_ * 16, 256), 256, 0, stream>>>(e0, ep0, eg0, BB, E0_, E1_, 128);
  run_mpl(temp0, stream, xg0, eg0, g1, BB, N1_, E1_, 128, 256, l0ske, l0skn, xs0, junkE);
  run_mpl(temp0, stream, x0, e0, g0, BB, N0_, E0_, 128, 128, l0m1e, l0m1n, x0, e0);
  gather_rows<<<cdiv((long)BB * N1_ * 16, 256), 256, 0, stream>>>(x0, m_id0, x1g, BB, N0_, N1_, 128);
  gather_rows<<<cdiv((long)BB * E1_ * 16, 256), 256, 0, stream>>>(e0, ep0, e1g, BB, E0_, E1_, 128);
  run_mpl(temp0, stream, x1g, e1g, g1, BB, N1_, E1_, 128, 256, l0m2e, l0m2n, x1, e1);
  residual_relu<<<cdiv(rN1 * 256, 256), 256, 0, stream>>>(x1, xs0, rN1 * 256);

  // ---- _res_down level 1 (g1 -> g2) ----
  gather_rows<<<cdiv((long)BB * N2_ * 32, 256), 256, 0, stream>>>(x1, m_id1, xg1, BB, N1_, N2_, 256);
  gather_rows<<<cdiv((long)BB * E2_ * 32, 256), 256, 0, stream>>>(e1, ep1, eg1, BB, E1_, E2_, 256);
  run_mpl(temp0, stream, xg1, eg1, g2, BB, N2_, E2_, 256, 512, l1ske, l1skn, xs1, junkE);
  run_mpl(temp0, stream, x1, e1, g1, BB, N1_, E1_, 256, 256, l1m1e, l1m1n, x1, e1);
  gather_rows<<<cdiv((long)BB * N2_ * 32, 256), 256, 0, stream>>>(x1, m_id1, x2g, BB, N1_, N2_, 256);
  gather_rows<<<cdiv((long)BB * E2_ * 32, 256), 256, 0, stream>>>(e1, ep1, e2g, BB, E1_, E2_, 256);
  run_mpl(temp0, stream, x2g, e2g, g2, BB, N2_, E2_, 256, 512, l1m2e, l1m2n, x2, e2b);
  residual_relu<<<cdiv((long)BB * N2_ * 512, 256), 256, 0, stream>>>(x2, xs1, (long)BB * N2_ * 512);

  // ---- bottom ----
  run_mpl(temp0, stream, x2, e2b, g2, BB, N2_, E2_, 512, 64, botE, botN, xb, junkE);

  // ---- VAE head ----
  head_transpose<<<cdiv(256 * 640, 256), 256, 0, stream>>>(xb, Ah);
  gemmRun(stream, Ah, pwHead, bmn, 256, 64, 640, 0, zx, nullptr);
  vae_head<<<1, 256, 0, stream>>>(zx, Wmu, bmu, Wlv, blv, eps, (float*)d_out);
}
